// ValleAR_48670569398813
// MI455X (gfx1250) — matrix operand order repaired and checked by reading, not compiled, not run on hardware
//
#include <hip/hip_runtime.h>
#include <stdint.h>

#define B_SZ   2
#define XL     256
#define CL     767
#define YL     768
#define SEQ    1024
#define DM     1024
#define FFD    4096
#define NL     6
#define NH     16
#define DH     64
#define NPROJ  1025
#define EOS_ID 1024
#define BOS_ID 1025

typedef __attribute__((ext_vector_type(16))) __bf16 v16bf;
typedef __attribute__((ext_vector_type(8)))  float  v8f;

struct __align__(16) U8x4 { unsigned int x[8]; };

// native RNE convert; lets the backend use v_cvt_*bf16* instructions
__device__ __forceinline__ unsigned short f2bf(float f) {
  __bf16 b = (__bf16)f;
  return __builtin_bit_cast(unsigned short, b);
}
__device__ __forceinline__ unsigned int pack2(unsigned short lo, unsigned short hi) {
  return (unsigned int)lo | ((unsigned int)hi << 16);
}

// A-operand fragment from LDS: 16 consecutive K values at p
__device__ __forceinline__ v16bf lds_frag_a(const unsigned short* p) {
  U8x4 t;
  *(uint4*)&t.x[0] = *(const uint4*)(p);
  *(uint4*)&t.x[4] = *(const uint4*)(p + 8);
  return __builtin_bit_cast(v16bf, t);
}
// B-operand fragment from LDS: 16 consecutive K values at p
__device__ __forceinline__ v16bf lds_frag_b(const unsigned short* p) {
  U8x4 t;
  *(uint4*)&t.x[0] = *(const uint4*)(p);
  *(uint4*)&t.x[4] = *(const uint4*)(p + 8);
  return __builtin_bit_cast(v16bf, t);
}
// A-operand fragment from a global fp32 row: 16 consecutive floats at rp
__device__ __forceinline__ v16bf row_frag_a(const float* rp) {
  float4 a = *(const float4*)(rp + 0);
  float4 b = *(const float4*)(rp + 4);
  float4 c = *(const float4*)(rp + 8);
  float4 d = *(const float4*)(rp + 12);
  U8x4 t;
  t.x[0]=pack2(f2bf(a.x),f2bf(a.y)); t.x[1]=pack2(f2bf(a.z),f2bf(a.w));
  t.x[2]=pack2(f2bf(b.x),f2bf(b.y)); t.x[3]=pack2(f2bf(b.z),f2bf(b.w));
  t.x[4]=pack2(f2bf(c.x),f2bf(c.y)); t.x[5]=pack2(f2bf(c.z),f2bf(c.w));
  t.x[6]=pack2(f2bf(d.x),f2bf(d.y)); t.x[7]=pack2(f2bf(d.z),f2bf(d.w));
  return __builtin_bit_cast(v16bf, t);
}
// B-operand fragment from a global fp32 row: 16 consecutive floats at rp
__device__ __forceinline__ v16bf row_frag_b(const float* rp) {
  float4 a = *(const float4*)(rp + 0);
  float4 b = *(const float4*)(rp + 4);
  float4 c = *(const float4*)(rp + 8);
  float4 d = *(const float4*)(rp + 12);
  U8x4 t;
  t.x[0]=pack2(f2bf(a.x),f2bf(a.y)); t.x[1]=pack2(f2bf(a.z),f2bf(a.w));
  t.x[2]=pack2(f2bf(b.x),f2bf(b.y)); t.x[3]=pack2(f2bf(b.z),f2bf(b.w));
  t.x[4]=pack2(f2bf(c.x),f2bf(c.y)); t.x[5]=pack2(f2bf(c.z),f2bf(c.w));
  t.x[6]=pack2(f2bf(d.x),f2bf(d.y)); t.x[7]=pack2(f2bf(d.z),f2bf(d.w));
  return __builtin_bit_cast(v16bf, t);
}
__device__ __forceinline__ v8f wmma_bf16(v16bf a, v16bf b, v8f c) {
  return __builtin_amdgcn_wmma_f32_16x16x32_bf16(false, a, false, b, (short)0, c,
                                                 false, false);
}

// ---------------------------------------------------------------------------
// Embedding + sinusoidal positional encoding -> h [B*SEQ, DM] fp32
// ---------------------------------------------------------------------------
__global__ __launch_bounds__(256) void embed_kernel(
    const int* __restrict__ tokens, const int* __restrict__ codes,
    const float* __restrict__ tok_emb, const float* __restrict__ audio_emb,
    float* __restrict__ h) {
  const int row = blockIdx.x;          // b*SEQ + s
  const int b = row / SEQ, s = row % SEQ;
  const float* src;
  int pos;
  if (s < XL) {
    src = tok_emb + (size_t)tokens[b * XL + s] * DM;
    pos = s;
  } else {
    int t = s - XL;
    int id = (t == 0) ? BOS_ID : codes[b * CL + (t - 1)];
    src = audio_emb + (size_t)id * DM;
    pos = t;
  }
  const float kln = 9.210340371976184f / (float)DM;  // ln(10000)/D
  const float fpos = (float)pos;
#pragma unroll
  for (int i = 0; i < 4; i++) {
    int d = threadIdx.x + i * 256;
    int p = d >> 1;
    float ang = fpos * __expf(-(float)(2 * p) * kln);
    float pe = (d & 1) ? __cosf(ang) : __sinf(ang);
    h[(size_t)row * DM + d] = src[d] + pe;
  }
}

// ---------------------------------------------------------------------------
// Generic bf16-WMMA GEMM:  C[M,N] = (relu?)(A[M,K] @ B[K,N] + bias)
// Block 256 thr / 8 waves, tile 128x128, K staged 32/step,
// double-buffered LDS + register prefetch: 1 barrier per stage, global-load
// latency overlapped with the 64 WMMAs of the previous stage.
// ---------------------------------------------------------------------------
__global__ __launch_bounds__(256) void gemm_kernel(
    const float* __restrict__ A, const float* __restrict__ Bm,
    const float* __restrict__ bias, float* __restrict__ C,
    int M, int N, int K, int lda, int ldb, int ldc, int relu) {
  __shared__ __align__(16) unsigned short As[2][128 * 40];  // [m][k], stride 40
  __shared__ __align__(16) unsigned short Bs[2][128 * 40];  // [n][k], stride 40
  const int tid = threadIdx.x;
  const int lane = tid & 31, wid = tid >> 5;
  const int l16 = lane & 15, lh = lane >> 4;
  const int wm = wid >> 1, wn = wid & 1;   // 4x2 wave grid, wave tile 32x64
  const int bm = blockIdx.y * 128;
  const int bn = blockIdx.x * 128;

  v8f zero = {0.f, 0.f, 0.f, 0.f, 0.f, 0.f, 0.f, 0.f};
  v8f acc[2][4];
#pragma unroll
  for (int i = 0; i < 2; i++)
#pragma unroll
    for (int j = 0; j < 4; j++) acc[i][j] = zero;

  float4 ra[4];      // staged A values (128x32 tile / 256 threads)
  float  rb[4][4];   // staged B values (32x128 tile / 256 threads)

  auto load_stage = [&](int k0) {
#pragma unroll
    for (int i = 0; i < 4; i++) {   // A: row = idx>>3, c4 = (idx&7)*4
      int idx = tid + 256 * i;
      int gr = bm + (idx >> 3); if (gr >= M) gr = M - 1;
      ra[i] = *(const float4*)(A + (size_t)gr * lda + k0 + (idx & 7) * 4);
    }
#pragma unroll
    for (int i = 0; i < 4; i++) {   // B: kk = idx>>5, n4 = (idx&31)*4
      int idx = tid + 256 * i;
      int kk = idx >> 5;
      int gn = bn + (idx & 31) * 4;
      const float* bp = Bm + (size_t)(k0 + kk) * ldb;
      if (((ldb & 3) == 0) && (gn + 3 < N)) {
        float4 f = *(const float4*)(bp + gn);
        rb[i][0] = f.x; rb[i][1] = f.y; rb[i][2] = f.z; rb[i][3] = f.w;
      } else {
#pragma unroll
        for (int j = 0; j < 4; j++) rb[i][j] = (gn + j < N) ? bp[gn + j] : 0.f;
      }
    }
  };

  const int nst = K >> 5;   // K multiple of 32 at every call site
  load_stage(0);

  for (int s = 0; s < nst; s++) {
    const int buf = s & 1;
    // commit staged registers to LDS (bf16)
#pragma unroll
    for (int i = 0; i < 4; i++) {
      int idx = tid + 256 * i;
      unsigned short* p = &As[buf][(idx >> 3) * 40 + (idx & 7) * 4];
      p[0] = f2bf(ra[i].x); p[1] = f2bf(ra[i].y);
      p[2] = f2bf(ra[i].z); p[3] = f2bf(ra[i].w);
    }
#pragma unroll
    for (int i = 0; i < 4; i++) {
      int idx = tid + 256 * i;
      int kk = idx >> 5;
      int n4 = (idx & 31) * 4;
#pragma unroll
      for (int j = 0; j < 4; j++) Bs[buf][(n4 + j) * 40 + kk] = f2bf(rb[i][j]);
    }
    __syncthreads();
    if (s + 1 < nst) load_stage((s + 1) * 32);  // overlap with WMMA below

    v16bf af[2], bfr[4];
#pragma unroll
    for (int mf = 0; mf < 2; mf++)
      af[mf] = lds_frag_a(&As[buf][(wm * 32 + mf * 16 + l16) * 40 + lh * 16]);
#pragma unroll
    for (int nf = 0; nf < 4; nf++)
      bfr[nf] = lds_frag_b(&Bs[buf][(wn * 64 + nf * 16 + l16) * 40 + lh * 16]);
#pragma unroll
    for (int mf = 0; mf < 2; mf++)
#pragma unroll
      for (int nf = 0; nf < 4; nf++)
        acc[mf][nf] = wmma_bf16(af[mf], bfr[nf], acc[mf][nf]);
  }
  // epilogue: bias (+ ReLU) + fp32 store
#pragma unroll
  for (int mf = 0; mf < 2; mf++) {
#pragma unroll
    for (int nf = 0; nf < 4; nf++) {
      int col = bn + wn * 64 + nf * 16 + l16;
      float bv = (bias != nullptr && col < N) ? bias[col] : 0.f;
#pragma unroll
      for (int r = 0; r < 8; r++) {
        int row = bm + wm * 32 + mf * 16 + r + 8 * lh;
        if (row < M && col < N) {
          float v = acc[mf][nf][r] + bv;
          if (relu) v = fmaxf(v, 0.f);
          C[(size_t)row * ldc + col] = v;
        }
      }
    }
  }
}

// ---------------------------------------------------------------------------
// Masked attention, WMMA for QK^T and PV. Block = 4 waves, one (b,h,i-tile64).
// qkv layout: [(b*SEQ+s), 3*DM] with Q at col h*64, K at 1024+h*64, V at 2048+h*64.
// ---------------------------------------------------------------------------
__global__ __launch_bounds__(128) void attn_kernel(const float* __restrict__ qkv,
                                                   float* __restrict__ attnout) {
  const int it = blockIdx.x, hh = blockIdx.y, bb = blockIdx.z;
  const int i0 = it * 64;
  const int tid = threadIdx.x, wid = tid >> 5, lane = tid & 31;
  const int l16 = lane & 15, lh = lane >> 4;
  const int iw = i0 + wid * 16;                 // this wave's 16 query rows
  const float scale = 0.125f;                   // 1/sqrt(64)
  __shared__ __align__(16) unsigned short Pl[4][16 * 32];

  const float* qbase = qkv + (size_t)bb * SEQ * 3 * DM + hh * DH;

  // Q fragments (A operand), dh split into two K=32 steps
  v16bf qf[2];
  {
    const float* qp = qbase + (size_t)(iw + l16) * (3 * DM);
    qf[0] = row_frag_a(qp + 0 + lh * 16);
    qf[1] = row_frag_a(qp + 32 + lh * 16);
  }
  // text rows attend [0,xl); audio rows attend j<=i (superset includes text)
  const int nj = (i0 < XL) ? XL : (i0 + 64);
  const int irow0 = iw + 8 * lh;

  // ---- pass 1: row max ----
  float rmax[8];
#pragma unroll
  for (int r = 0; r < 8; r++) rmax[r] = -3.0e38f;
  for (int jb = 0; jb < nj; jb += 16) {
    v8f sc = {0.f, 0.f, 0.f, 0.f, 0.f, 0.f, 0.f, 0.f};
    const float* kp = qbase + DM + (size_t)(jb + l16) * (3 * DM);
    sc = wmma_bf16(qf[0], row_frag_b(kp + 0 + lh * 16), sc);
    sc = wmma_bf16(qf[1], row_frag_b(kp + 32 + lh * 16), sc);
    const int j = jb + l16;
#pragma unroll
    for (int r = 0; r < 8; r++) {
      float s = sc[r] * scale;
      int i = irow0 + r;
      bool ok = (i < XL) ? (j < XL) : (j <= i);
      if (!ok) s = -3.0e38f;
      rmax[r] = fmaxf(rmax[r], s);
    }
  }
#pragma unroll
  for (int r = 0; r < 8; r++)
#pragma unroll
    for (int off = 1; off < 16; off <<= 1)
      rmax[r] = fmaxf(rmax[r], __shfl_xor(rmax[r], off));

  // ---- pass 2: exp, row sum, P@V ----
  float rsum[8];
#pragma unroll
  for (int r = 0; r < 8; r++) rsum[r] = 0.f;
  v8f zero = {0.f, 0.f, 0.f, 0.f, 0.f, 0.f, 0.f, 0.f};
  v8f oacc[4];
#pragma unroll
  for (int nf = 0; nf < 4; nf++) oacc[nf] = zero;
  unsigned short* myP = &Pl[wid][0];

  for (int jb = 0; jb < nj; jb += 32) {
#pragma unroll
    for (int half = 0; half < 2; half++) {
      v8f sc = {0.f, 0.f, 0.f, 0.f, 0.f, 0.f, 0.f, 0.f};
      const float* kp = qbase + DM + (size_t)(jb + half * 16 + l16) * (3 * DM);
      sc = wmma_bf16(qf[0], row_frag_b(kp + 0 + lh * 16), sc);
      sc = wmma_bf16(qf[1], row_frag_b(kp + 32 + lh * 16), sc);
      const int j = jb + half * 16 + l16;
#pragma unroll
      for (int r = 0; r < 8; r++) {
        float s = sc[r] * scale;
        int i = irow0 + r;
        bool ok = (i < XL) ? (j < XL) : (j <= i);
        float p = ok ? __expf(s - rmax[r]) : 0.f;
        rsum[r] += p;
        myP[(r + 8 * lh) * 32 + half * 16 + l16] = f2bf(p);  // P tile bf16
      }
    }
    __syncthreads();
    v16bf pf = lds_frag_a(myP + l16 * 32 + lh * 16);  // A operand (16x32)
    const float* vb = qbase + 2 * DM + (size_t)jb * (3 * DM);
#pragma unroll
    for (int nf = 0; nf < 4; nf++) {
      U8x4 t;
      const int c = nf * 16 + l16;
#pragma unroll
      for (int w2 = 0; w2 < 8; w2++) {
        int jj = lh * 16 + 2 * w2;  // B operand K order: 16*half + e
        float f0 = vb[(size_t)jj * (3 * DM) + c];
        float f1 = vb[(size_t)(jj + 1) * (3 * DM) + c];
        t.x[w2] = pack2(f2bf(f0), f2bf(f1));
      }
      v16bf vf = __builtin_bit_cast(v16bf, t);
      oacc[nf] = wmma_bf16(pf, vf, oacc[nf]);
    }
    __syncthreads();
  }
#pragma unroll
  for (int r = 0; r < 8; r++)
#pragma unroll
    for (int off = 1; off < 16; off <<= 1)
      rsum[r] += __shfl_xor(rsum[r], off);
#pragma unroll
  for (int nf = 0; nf < 4; nf++)
#pragma unroll
    for (int r = 0; r < 8; r++) {
      float o = oacc[nf][r] / rsum[r];
      int srow = iw + r + 8 * lh;
      attnout[((size_t)bb * SEQ + srow) * DM + hh * DH + nf * 16 + l16] = o;
    }
}

// ---------------------------------------------------------------------------
// h = LayerNorm(h + add) * w + b   (one block per row, D=1024)
// ---------------------------------------------------------------------------
__global__ __launch_bounds__(256) void ln_residual_kernel(
    float* __restrict__ h, const float* __restrict__ add,
    const float* __restrict__ w, const float* __restrict__ b) {
  __shared__ float s1[256], s2[256];
  const int row = blockIdx.x, tid = threadIdx.x;
  const size_t base = (size_t)row * DM;
  float loc[4], a = 0.f, q = 0.f;
#pragma unroll
  for (int i = 0; i < 4; i++) {
    int c = tid + i * 256;
    float v = h[base + c] + add[base + c];
    loc[i] = v; a += v; q += v * v;
  }
  s1[tid] = a; s2[tid] = q; __syncthreads();
  for (int off = 128; off > 0; off >>= 1) {
    if (tid < off) { s1[tid] += s1[tid + off]; s2[tid] += s2[tid + off]; }
    __syncthreads();
  }
  float mu = s1[0] / (float)DM;
  float var = s2[0] / (float)DM - mu * mu;
  float inv = rsqrtf(var + 1e-5f);
#pragma unroll
  for (int i = 0; i < 4; i++) {
    int c = tid + i * 256;
    h[base + c] = (loc[i] - mu) * inv * w[c] + b[c];
  }
}

// Gather audio rows + final LayerNorm -> hf [B*YL, DM]
__global__ __launch_bounds__(256) void gather_lnf_kernel(
    const float* __restrict__ h, float* __restrict__ hf,
    const float* __restrict__ w, const float* __restrict__ b) {
  __shared__ float s1[256], s2[256];
  const int ro = blockIdx.x, tid = threadIdx.x;
  const int bb = ro / YL, t = ro % YL;
  const size_t base = ((size_t)bb * SEQ + XL + t) * DM;
  float loc[4], a = 0.f, q = 0.f;
#pragma unroll
  for (int i = 0; i < 4; i++) {
    int c = tid + i * 256;
    float v = h[base + c];
    loc[i] = v; a += v; q += v * v;
  }
  s1[tid] = a; s2[tid] = q; __syncthreads();
  for (int off = 128; off > 0; off >>= 1) {
    if (tid < off) { s1[tid] += s1[tid + off]; s2[tid] += s2[tid + off]; }
    __syncthreads();
  }
  float mu = s1[0] / (float)DM;
  float var = s2[0] / (float)DM - mu * mu;
  float inv = rsqrtf(var + 1e-5f);
#pragma unroll
  for (int i = 0; i < 4; i++) {
    int c = tid + i * 256;
    hf[(size_t)ro * DM + c] = (loc[i] - mu) * inv * w[c] + b[c];
  }
}

// per-row NLL: logsumexp(logits) - logits[target]
__global__ __launch_bounds__(256) void nll_kernel(const float* __restrict__ logits,
                                                  const int* __restrict__ codes,
                                                  float* __restrict__ rownll) {
  __shared__ float sm[256];
  const int ro = blockIdx.x, tid = threadIdx.x;
  const float* L = logits + (size_t)ro * NPROJ;
  float m = -3.0e38f;
  for (int c = tid; c < NPROJ; c += 256) m = fmaxf(m, L[c]);
  sm[tid] = m; __syncthreads();
  for (int off = 128; off > 0; off >>= 1) {
    if (tid < off) sm[tid] = fmaxf(sm[tid], sm[tid + off]);
    __syncthreads();
  }
  m = sm[0]; __syncthreads();
  float s = 0.f;
  for (int c = tid; c < NPROJ; c += 256) s += __expf(L[c] - m);
  sm[tid] = s; __syncthreads();
  for (int off = 128; off > 0; off >>= 1) {
    if (tid < off) sm[tid] += sm[tid + off];
    __syncthreads();
  }
  if (tid == 0) {
    int bb = ro / YL, t = ro % YL;
    int tgt = (t < CL) ? codes[bb * CL + t] : EOS_ID;
    rownll[ro] = (m + logf(sm[0])) - L[tgt];
  }
}

__global__ __launch_bounds__(256) void reduce_mean_kernel(const float* __restrict__ v,
                                                          float* __restrict__ out,
                                                          int n) {
  __shared__ float sm[256];
  float s = 0.f;
  for (int i = threadIdx.x; i < n; i += 256) s += v[i];
  sm[threadIdx.x] = s; __syncthreads();
  for (int off = 128; off > 0; off >>= 1) {
    if (threadIdx.x < off) sm[threadIdx.x] += sm[threadIdx.x + off];
    __syncthreads();
  }
  if (threadIdx.x == 0) out[0] = sm[0] / (float)n;
}

// ---------------------------------------------------------------------------
static inline void launch_gemm(const float* A, const float* B, const float* bias,
                               float* C, int M, int N, int K, int lda, int ldb,
                               int ldc, int relu, hipStream_t s) {
  dim3 g((N + 127) / 128, M / 128);
  gemm_kernel<<<g, 256, 0, s>>>(A, B, bias, C, M, N, K, lda, ldb, ldc, relu);
}

extern "C" void kernel_launch(void* const* d_in, const int* in_sizes, int n_in,
                              void* d_out, int out_size, void* d_ws, size_t ws_size,
                              hipStream_t stream) {
  const int*   tokens    = (const int*)d_in[0];
  const int*   codes     = (const int*)d_in[1];
  const float* tok_emb   = (const float*)d_in[2];
  const float* audio_emb = (const float*)d_in[3];
  const float* Wqkv      = (const float*)d_in[4];
  const float* b_qkv     = (const float*)d_in[5];
  const float* Wo        = (const float*)d_in[6];
  const float* b_o       = (const float*)d_in[7];
  const float* W1        = (const float*)d_in[8];
  const float* b1        = (const float*)d_in[9];
  const float* W2        = (const float*)d_in[10];
  const float* b2        = (const float*)d_in[11];
  const float* ln1w      = (const float*)d_in[12];
  const float* ln1b      = (const float*)d_in[13];
  const float* ln2w      = (const float*)d_in[14];
  const float* ln2b      = (const float*)d_in[15];
  const float* lnfw      = (const float*)d_in[16];
  const float* lnfb      = (const float*)d_in[17];
  const float* projw     = (const float*)d_in[18];
  float* out = (float*)d_out;

  const int ROWS = B_SZ * SEQ;      // 2048
  const int PROWS = B_SZ * YL;      // 1536
  float* h      = (float*)d_ws;
  float* qkvb   = h + (size_t)ROWS * DM;
  float* attnb  = qkvb + (size_t)ROWS * 3 * DM;
  float* tmp    = attnb + (size_t)ROWS * DM;
  float* ff1    = tmp + (size_t)ROWS * DM;
  float* hf     = ff1 + (size_t)ROWS * FFD;
  float* logits = hf + (size_t)PROWS * DM;
  float* rownll = logits + (size_t)PROWS * NPROJ;
  (void)ws_size; (void)in_sizes; (void)n_in; (void)out_size;

  embed_kernel<<<ROWS, 256, 0, stream>>>(tokens, codes, tok_emb, audio_emb, h);

  for (int l = 0; l < NL; l++) {
    const float* wq = Wqkv + (size_t)l * DM * 3 * DM;
    const float* bq = b_qkv + (size_t)l * 3 * DM;
    const float* wo = Wo + (size_t)l * DM * DM;
    const float* bo = b_o + (size_t)l * DM;
    const float* w1 = W1 + (size_t)l * DM * FFD;
    const float* bb1 = b1 + (size_t)l * FFD;
    const float* w2 = W2 + (size_t)l * FFD * DM;
    const float* bb2 = b2 + (size_t)l * DM;

    launch_gemm(h, wq, bq, qkvb, ROWS, 3 * DM, DM, DM, 3 * DM, 3 * DM, 0, stream);
    attn_kernel<<<dim3(SEQ / 64, NH, B_SZ), 128, 0, stream>>>(qkvb, attnb);
    launch_gemm(attnb, wo, bo, tmp, ROWS, DM, DM, DM, DM, DM, 0, stream);
    ln_residual_kernel<<<ROWS, 256, 0, stream>>>(h, tmp, ln1w + (size_t)l * DM,
                                                 ln1b + (size_t)l * DM);
    launch_gemm(h, w1, bb1, ff1, ROWS, FFD, DM, DM, FFD, FFD, 1, stream);
    launch_gemm(ff1, w2, bb2, tmp, ROWS, DM, FFD, FFD, DM, DM, 0, stream);
    ln_residual_kernel<<<ROWS, 256, 0, stream>>>(h, tmp, ln2w + (size_t)l * DM,
                                                 ln2b + (size_t)l * DM);
  }

  gather_lnf_kernel<<<PROWS, 256, 0, stream>>>(h, hf, lnfw, lnfb);
  launch_gemm(hf, projw, nullptr, logits, PROWS, NPROJ, DM, DM, NPROJ, NPROJ, 0,
              stream);
  nll_kernel<<<PROWS, 256, 0, stream>>>(logits, codes, rownll);
  reduce_mean_kernel<<<1, 256, 0, stream>>>(rownll, out, PROWS);
}
